// EGCN_24094766531069
// MI455X (gfx1250) — matrix operand order repaired and checked by reading, not compiled, not run on hardware
//
#include <hip/hip_runtime.h>

// ---------------------------------------------------------------------------
// EGNN layer for MI455X (gfx1250, wave32, WMMA).
// Compute-bound (~71 GFLOP, 93% in per-edge 64-wide MLPs); weights + node
// tables + scatter accumulators are L2-resident (192MB L2), so we spend the
// effort on the matrix path: v_wmma_f32_16x16x32_bf16 (f32 accumulate).
// One wave owns a 16-row tile (16 edges / 16 nodes); activations ping-pong
// through wave-private LDS (same-wave DS ordering => no barriers needed).
// ---------------------------------------------------------------------------

typedef __attribute__((ext_vector_type(16))) __bf16        v16bf;
typedef __attribute__((ext_vector_type(8)))  float         v8f;
typedef __attribute__((ext_vector_type(4)))  unsigned int  v4u;

struct F32B { v4u a, b; };   // 32 bytes == one v16bf fragment

__device__ __forceinline__ unsigned short f2bf(float x) {
  unsigned int u = __float_as_uint(x);
  u += 0x7FFFu + ((u >> 16) & 1u);          // round-to-nearest-even
  return (unsigned short)(u >> 16);
}
__device__ __forceinline__ float bf2f(unsigned short s) {
  return __uint_as_float(((unsigned int)s) << 16);
}

// B fragment: packed weights laid out by prep kernel so each lane reads
// 32 contiguous bytes:  w[(kt*4+nt)*512 + lane*16 + j] = W[k][n],
//   k = kt*32 + (j<8 ? (lane>>4)*8+j : 16+(lane>>4)*8+(j-8)),  n = nt*16 + (lane&15)
__device__ __forceinline__ v16bf ld_b_frag(const unsigned short* p) {
  F32B f;
  f.a = *reinterpret_cast<const v4u*>(p);
  f.b = *reinterpret_cast<const v4u*>(p + 8);
  return __builtin_bit_cast(v16bf, f);
}

// A fragment (16x32 bf16, ISA layout): lane row m=lane&15, h=lane>>4,
// element j: k = (j<8) ? h*8+j : 16+h*8+(j-8)  -> two 16B runs per lane.
template<int LDK>
__device__ __forceinline__ v16bf ld_a_frag(const unsigned short* act, int koff, int lane) {
  int m = lane & 15, h = lane >> 4;
  const unsigned short* r0 = act + m * LDK + koff + h * 8;
  F32B f;
  f.a = *reinterpret_cast<const v4u*>(r0);
  f.b = *reinterpret_cast<const v4u*>(r0 + 16);
  return __builtin_bit_cast(v16bf, f);
}

// One 16xK -> 16x64 GEMM tile: 4 N-tiles x KC K-chunks of WMMA.
// bias is always non-null (branch-free hot loop).
template<int LDK, int KC>
__device__ __forceinline__ void gemm16(const unsigned short* act,
                                       const unsigned short* w,
                                       const float* __restrict__ bias,
                                       v8f* c, int lane) {
  int col = lane & 15;
  #pragma unroll
  for (int nt = 0; nt < 4; ++nt) {
    float bv = bias[nt * 16 + col];
    v8f acc = { bv, bv, bv, bv, bv, bv, bv, bv };
    #pragma unroll
    for (int kt = 0; kt < KC; ++kt) {
      v16bf a = ld_a_frag<LDK>(act, kt * 32, lane);
      v16bf b = ld_b_frag(w + (kt * 4 + nt) * 512 + lane * 16);
      acc = __builtin_amdgcn_wmma_f32_16x16x32_bf16(false, a, false, b,
                                                    (short)0, acc, false, false);
    }
    c[nt] = acc;
  }
}

// Store D fragments (row = r + (lane>>4)*8, col = nt*16 + lane&15) to LDS bf16.
template<int LDK, bool LEAKY>
__device__ __forceinline__ void store_act(unsigned short* dst, const v8f* c, int lane) {
  int col = lane & 15, h = lane >> 4;
  #pragma unroll
  for (int nt = 0; nt < 4; ++nt)
    #pragma unroll
    for (int r = 0; r < 8; ++r) {
      float v = c[nt][r];
      if (LEAKY) v = (v >= 0.f) ? v : 0.01f * v;
      dst[(h * 8 + r) * LDK + nt * 16 + col] = f2bf(v);
    }
}

// Packed-weight offsets (bf16 elements) inside d_ws.
enum : int {
  OFF_E0 = 0,      OFF_E1 = 10240, OFF_E2 = 14336, OFF_E3 = 18432, OFF_E4 = 22528,
  OFF_C0 = 26624,  OFF_C1 = 30720, OFF_C2 = 34816, OFF_C3 = 38912,
  OFF_N0 = 43008,  OFF_N1 = 51200, OFF_N2 = 55296, OFF_N3 = 59392, OFF_N4 = 63488,
  OFF_V0 = 67584,  OFF_V1 = 71680, OFF_V2 = 75776, OFF_V3 = 79872,
  OFF_LH = 83968,  WPK_ELEMS = 88064
};

#define EW 4   // waves per block (128 threads)

// ---------------------------------------------------------------------------
// Weight packing: f32 (K_src x 64, row-major) -> bf16 fragment order.
// ---------------------------------------------------------------------------
__global__ void prep_pack(const float* __restrict__ W, unsigned short* __restrict__ dst,
                          int ktiles, int ksrc) {
  int idx = blockIdx.x * blockDim.x + threadIdx.x;
  int total = ktiles * 4 * 512;
  if (idx >= total) return;
  int kt   = idx / 2048;
  int rem  = idx - kt * 2048;
  int nt   = rem / 512;
  int q    = rem - nt * 512;
  int lane = q >> 4, j = q & 15;
  int k = kt * 32 + ((j < 8) ? (lane >> 4) * 8 + j : 16 + (lane >> 4) * 8 + (j - 8));
  int n = nt * 16 + (lane & 15);
  float v = (k < ksrc) ? W[k * 64 + n] : 0.f;
  dst[idx] = f2bf(v);
}

// ---------------------------------------------------------------------------
// Edge pipeline: gather -> edge_mlp (5) -> coord_mlp (4+head) -> scatter.
// ---------------------------------------------------------------------------
struct EdgeArgs {
  const float* __restrict__ nf; const float* __restrict__ coord;
  const float* __restrict__ ef;
  const int* __restrict__ src; const int* __restrict__ dst;
  const unsigned short* __restrict__ wpk;
  const float *eb0, *eb1, *eb2, *eb3, *eb4;
  const float *cb0, *cb1, *cb2, *cb3;
  const float* __restrict__ cw4;
  float* __restrict__ sum_x; float* __restrict__ deg; float* __restrict__ h_nei;
  int mTiles;
};

__global__ __launch_bounds__(EW * 32) void egnn_edge(EdgeArgs A) {
  __shared__ __align__(16) unsigned short sB0[EW][16][160];
  __shared__ __align__(16) unsigned short sB1[EW][16][64];
  __shared__ __align__(16) unsigned short sMH[EW][16][64];
  __shared__ int sd[EW][16];

  int wave = threadIdx.x >> 5, lane = threadIdx.x & 31;
  int tile = blockIdx.x * EW + wave;
  if (tile >= A.mTiles) return;
  int e0 = tile * 16;

  unsigned short* B0 = &sB0[wave][0][0];
  unsigned short* B1 = &sB1[wave][0][0];
  unsigned short* MH = &sMH[wave][0][0];

  // per-edge geometry (lanes 0..15 own one edge each)
  float xd0 = 0.f, xd1 = 0.f;
  if (lane < 16) {
    int e = e0 + lane;
    int s = A.src[e], d = A.dst[e];
    sd[wave][lane] = d;
    float ax = A.coord[2 * s], ay = A.coord[2 * s + 1];
    float bx = A.coord[2 * d], by = A.coord[2 * d + 1];
    xd0 = ax - bx; xd1 = ay - by;
    float rad = xd0 * xd0 + xd1 * xd1;
    float inv = 1.f / (sqrtf(rad) + 1e-30f);
    xd0 *= inv; xd1 *= inv;
  }

  // Stage input f = [nf[src](64) | nf[dst](64) | radial | ef(8) | pad] -> 16x160
  for (int idx = lane; idx < 16 * 160; idx += 32) {
    int e = idx / 160, k = idx - e * 160;
    int eg = e0 + e;
    float v = 0.f;
    if (k < 64)        v = A.nf[(long)A.src[eg] * 64 + k];
    else if (k < 128)  v = A.nf[(long)A.dst[eg] * 64 + (k - 64)];
    else if (k == 128) {
      int s = A.src[eg], d = A.dst[eg];
      float dx = A.coord[2 * s] - A.coord[2 * d];
      float dy = A.coord[2 * s + 1] - A.coord[2 * d + 1];
      v = dx * dx + dy * dy;
    } else if (k < 137) v = A.ef[(long)eg * 8 + (k - 129)];
    B0[e * 160 + k] = f2bf(v);
  }

  v8f c[4];
  // edge_mlp: 137(->160) -> 64 -> 64 -> 64 -> 64 -> 64
  gemm16<160, 5>(B0, A.wpk + OFF_E0, A.eb0, c, lane); store_act<64, true>(B1, c, lane);
  gemm16<64, 2>(B1, A.wpk + OFF_E1, A.eb1, c, lane); store_act<160, true>(B0, c, lane);
  gemm16<160, 2>(B0, A.wpk + OFF_E2, A.eb2, c, lane); store_act<64, true>(B1, c, lane);
  gemm16<64, 2>(B1, A.wpk + OFF_E3, A.eb3, c, lane); store_act<160, true>(B0, c, lane);
  gemm16<160, 2>(B0, A.wpk + OFF_E4, A.eb4, c, lane);   // msg_h, no activation

  // scatter msg_h (f32 atomics into L2-resident accumulator) + keep bf16 copy
  {
    int col = lane & 15, h = lane >> 4;
    #pragma unroll
    for (int nt = 0; nt < 4; ++nt)
      #pragma unroll
      for (int r = 0; r < 8; ++r) {
        float v = c[nt][r];
        int row = h * 8 + r;
        MH[row * 64 + nt * 16 + col] = f2bf(v);
        atomicAdd(&A.h_nei[(long)sd[wave][row] * 64 + nt * 16 + col], v);
      }
  }

  // coord_mlp hidden: 64 -> 64 x4 (leaky)
  gemm16<64, 2>(MH, A.wpk + OFF_C0, A.cb0, c, lane); store_act<160, true>(B0, c, lane);
  gemm16<160, 2>(B0, A.wpk + OFF_C1, A.cb1, c, lane); store_act<64, true>(B1, c, lane);
  gemm16<64, 2>(B1, A.wpk + OFF_C2, A.cb2, c, lane); store_act<160, true>(B0, c, lane);
  gemm16<160, 2>(B0, A.wpk + OFF_C3, A.cb3, c, lane); store_act<64, true>(B1, c, lane);

  // coord head 64->1 (no bias) + msg_x scatter + degree
  if (lane < 16) {
    float s = 0.f;
    #pragma unroll
    for (int k = 0; k < 64; ++k) s += bf2f(B1[lane * 64 + k]) * A.cw4[k];
    int d = sd[wave][lane];
    atomicAdd(&A.sum_x[2 * d],     s * xd0);
    atomicAdd(&A.sum_x[2 * d + 1], s * xd1);
    atomicAdd(&A.deg[d], 1.f);
  }
}

// ---------------------------------------------------------------------------
// Node pipeline: vel_mlp, node_mlp, linh, coordinate update -> out (N x 66).
// ---------------------------------------------------------------------------
struct NodeArgs {
  const float* __restrict__ nf;
  const unsigned short* __restrict__ wpk;
  const float *nb0, *nb1, *nb2, *nb3, *nb4;
  const float *vb0, *vb1, *vb2, *vb3;
  const float* __restrict__ vw4;
  const float* __restrict__ linh_b;
  const float* __restrict__ sum_x; const float* __restrict__ deg;
  const float* __restrict__ h_nei;
  float* __restrict__ out;
  int nTiles;
};

__global__ __launch_bounds__(EW * 32) void egnn_node(NodeArgs A) {
  __shared__ __align__(16) unsigned short sB0[EW][16][128];
  __shared__ __align__(16) unsigned short sB1[EW][16][64];
  __shared__ __align__(16) unsigned short sB2[EW][16][64];

  int wave = threadIdx.x >> 5, lane = threadIdx.x & 31;
  int tile = blockIdx.x * EW + wave;
  if (tile >= A.nTiles) return;
  int n0 = tile * 16;

  unsigned short* B0 = &sB0[wave][0][0];
  unsigned short* B1 = &sB1[wave][0][0];
  unsigned short* B2 = &sB2[wave][0][0];

  // Stage [node_feat | h_neigh] -> 16x128
  for (int idx = lane; idx < 16 * 128; idx += 32) {
    int r = idx >> 7, k = idx & 127;
    long n = n0 + r;
    float v = (k < 64) ? A.nf[n * 64 + k] : A.h_nei[n * 64 + (k - 64)];
    B0[r * 128 + k] = f2bf(v);
  }

  v8f c[4];
  // vel_mlp on node_feat (cols 0..63 of B0)
  gemm16<128, 2>(B0, A.wpk + OFF_V0, A.vb0, c, lane); store_act<64, true>(B1, c, lane);
  gemm16<64, 2>(B1, A.wpk + OFF_V1, A.vb1, c, lane); store_act<64, true>(B2, c, lane);
  gemm16<64, 2>(B2, A.wpk + OFF_V2, A.vb2, c, lane); store_act<64, true>(B1, c, lane);
  gemm16<64, 2>(B1, A.wpk + OFF_V3, A.vb3, c, lane); store_act<64, true>(B2, c, lane);

  float vel = 0.f;
  if (lane < 16) {
    #pragma unroll
    for (int k = 0; k < 64; ++k) vel += bf2f(B2[lane * 64 + k]) * A.vw4[k];
  }

  // node_mlp: 128 -> 64 x5 (last layer no activation)
  gemm16<128, 4>(B0, A.wpk + OFF_N0, A.nb0, c, lane); store_act<64, true>(B1, c, lane);
  gemm16<64, 2>(B1, A.wpk + OFF_N1, A.nb1, c, lane); store_act<64, true>(B2, c, lane);
  gemm16<64, 2>(B2, A.wpk + OFF_N2, A.nb2, c, lane); store_act<64, true>(B1, c, lane);
  gemm16<64, 2>(B1, A.wpk + OFF_N3, A.nb3, c, lane); store_act<64, true>(B2, c, lane);
  gemm16<64, 2>(B2, A.wpk + OFF_N4, A.nb4, c, lane); store_act<64, false>(B1, c, lane);

  // linh: h @ W + b, written straight from fragments to out[:, 2:66]
  gemm16<64, 2>(B1, A.wpk + OFF_LH, A.linh_b, c, lane);
  {
    int col = lane & 15, h = lane >> 4;
    #pragma unroll
    for (int nt = 0; nt < 4; ++nt)
      #pragma unroll
      for (int r = 0; r < 8; ++r) {
        long n = n0 + h * 8 + r;
        A.out[n * 66 + 2 + nt * 16 + col] = c[nt][r];
      }
  }

  // out[:, 0:2] = sum_x/max(deg,1) + vel * node_feat[:, 0:2]
  if (lane < 16) {
    long n = n0 + lane;
    float dg = A.deg[n]; dg = (dg > 1.f) ? dg : 1.f;
    float v0 = A.nf[n * 64 + 0], v1 = A.nf[n * 64 + 1];
    A.out[n * 66 + 0] = A.sum_x[2 * n]     / dg + vel * v0;
    A.out[n * 66 + 1] = A.sum_x[2 * n + 1] / dg + vel * v1;
  }
}

// ---------------------------------------------------------------------------
extern "C" void kernel_launch(void* const* d_in, const int* in_sizes, int n_in,
                              void* d_out, int out_size, void* d_ws, size_t ws_size,
                              hipStream_t stream) {
  (void)n_in; (void)out_size; (void)ws_size;
  const int N = in_sizes[0] / 64;   // 50000
  const int M = in_sizes[3];        // 800000

  // d_ws layout: [packed bf16 weights | sum_x (N*2) | deg (N) | h_neigh (N*64)]
  unsigned short* wpk = (unsigned short*)d_ws;
  float* accum = (float*)((char*)d_ws + (size_t)WPK_ELEMS * 2);
  float* sum_x = accum;
  float* deg   = sum_x + (size_t)2 * N;
  float* h_nei = deg + N;
  hipMemsetAsync(accum, 0, (size_t)(3 * N + 64 * N) * sizeof(float), stream);

  // Pack all MLP weights into WMMA-B fragment order (bf16).
  struct PD { const float* W; int off, kt, ks; };
  const PD pd[19] = {
    { (const float*)d_in[5],  OFF_E0, 5, 137 },
    { (const float*)d_in[7],  OFF_E1, 2, 64 },
    { (const float*)d_in[9],  OFF_E2, 2, 64 },
    { (const float*)d_in[11], OFF_E3, 2, 64 },
    { (const float*)d_in[13], OFF_E4, 2, 64 },
    { (const float*)d_in[25], OFF_C0, 2, 64 },
    { (const float*)d_in[27], OFF_C1, 2, 64 },
    { (const float*)d_in[29], OFF_C2, 2, 64 },
    { (const float*)d_in[31], OFF_C3, 2, 64 },
    { (const float*)d_in[15], OFF_N0, 4, 128 },
    { (const float*)d_in[17], OFF_N1, 2, 64 },
    { (const float*)d_in[19], OFF_N2, 2, 64 },
    { (const float*)d_in[21], OFF_N3, 2, 64 },
    { (const float*)d_in[23], OFF_N4, 2, 64 },
    { (const float*)d_in[34], OFF_V0, 2, 64 },
    { (const float*)d_in[36], OFF_V1, 2, 64 },
    { (const float*)d_in[38], OFF_V2, 2, 64 },
    { (const float*)d_in[40], OFF_V3, 2, 64 },
    { (const float*)d_in[43], OFF_LH, 2, 64 },
  };
  for (int i = 0; i < 19; ++i) {
    int total = pd[i].kt * 4 * 512;
    prep_pack<<<(total + 255) / 256, 256, 0, stream>>>(pd[i].W, wpk + pd[i].off,
                                                       pd[i].kt, pd[i].ks);
  }

  EdgeArgs ea;
  ea.nf = (const float*)d_in[0]; ea.coord = (const float*)d_in[1];
  ea.ef = (const float*)d_in[2];
  ea.src = (const int*)d_in[3]; ea.dst = (const int*)d_in[4];
  ea.wpk = wpk;
  ea.eb0 = (const float*)d_in[6];  ea.eb1 = (const float*)d_in[8];
  ea.eb2 = (const float*)d_in[10]; ea.eb3 = (const float*)d_in[12];
  ea.eb4 = (const float*)d_in[14];
  ea.cb0 = (const float*)d_in[26]; ea.cb1 = (const float*)d_in[28];
  ea.cb2 = (const float*)d_in[30]; ea.cb3 = (const float*)d_in[32];
  ea.cw4 = (const float*)d_in[33];
  ea.sum_x = sum_x; ea.deg = deg; ea.h_nei = h_nei;
  ea.mTiles = M / 16;
  int egrid = (ea.mTiles + EW - 1) / EW;
  egnn_edge<<<egrid, EW * 32, 0, stream>>>(ea);

  NodeArgs na;
  na.nf = (const float*)d_in[0];
  na.wpk = wpk;
  na.nb0 = (const float*)d_in[16]; na.nb1 = (const float*)d_in[18];
  na.nb2 = (const float*)d_in[20]; na.nb3 = (const float*)d_in[22];
  na.nb4 = (const float*)d_in[24];
  na.vb0 = (const float*)d_in[35]; na.vb1 = (const float*)d_in[37];
  na.vb2 = (const float*)d_in[39]; na.vb3 = (const float*)d_in[41];
  na.vw4 = (const float*)d_in[42];
  na.linh_b = (const float*)d_in[44];
  na.sum_x = sum_x; na.deg = deg; na.h_nei = h_nei;
  na.out = (float*)d_out;
  na.nTiles = (N + 15) / 16;
  int ngrid = (na.nTiles + EW - 1) / EW;
  egnn_node<<<ngrid, EW * 32, 0, stream>>>(na);
}
